// SINQLinear_49761491091553
// MI455X (gfx1250) — compile-verified, matrix operand order checked by reading, not run on hardware
//
#include <hip/hip_runtime.h>
#include <hip/hip_bf16.h>

// SINQ 4-bit dequant linear: out[16,8192] = x[16,8192] . W^T + bias
// W = (W_q - zero[g]) * scale[g] * s2[row], group size 64 along IN.
//
// Memory-bound: W_q is 256 MB of int32 codes -> ~11us floor at 23.3 TB/s.
// Strategy: stream W_q with b128 loads, dequant via one v_fma per element
// (a = scale*s2, b = -zero*scale*s2), pack to bf16, and accumulate with
// v_wmma_f32_16x16x32_bf16 (M=16 tokens fits one tile). 8 waves per block
// split K=8192 8 ways; LDS reduction combines partials.

typedef __attribute__((ext_vector_type(16))) __bf16 v16bf;
typedef __attribute__((ext_vector_type(8)))  float  v8f;
typedef __attribute__((ext_vector_type(4)))  int    v4i;
typedef __attribute__((ext_vector_type(4)))  float  v4f;

#define OUTF   8192
#define INF    8192
#define NGROUP 128
#define GSIZE  64      // IN / GROUPS
#define NTOK   16
#define WAVES  8       // waves per block (256 threads, wave32)
#define KSLICE (INF / WAVES)   // 1024 K per wave = 16 groups = 32 WMMAs

__global__ __launch_bounds__(256) void sinq_wmma_kernel(
    const float* __restrict__ x,      // [16, 8192]
    const int*   __restrict__ Wq,     // [8192, 8192] codes 0..15 as int32
    const float* __restrict__ scale,  // [8192, 128]
    const float* __restrict__ zero,   // [8192, 128]
    const float* __restrict__ s2,     // [8192, 1]
    const float* __restrict__ bias,   // [8192]
    float*       __restrict__ out)    // [16, 8192]
{
    __shared__ float red[WAVES][32][9];   // padded: stride 9 avoids 64-bank conflicts

    const int lane  = threadIdx.x & 31;
    const int wave  = threadIdx.x >> 5;
    const int obase = blockIdx.x * 16;    // 16 output features per block
    const int n     = lane & 15;          // B: output column; A: token row
    const int half  = lane >> 4;          // K-half selector per ISA layout

    const int  row = obase + n;           // weight row this lane feeds into B
    const float s2r = s2[row];

    const int k_begin = wave * KSLICE;
    const int g_begin = k_begin / GSIZE;

    const int*   wrow = Wq + (long)row * INF;
    const float* xrow = x  + (long)n   * INF;   // token n (A lanes 16-31 reuse same M)

    v8f acc = {};

    for (int g = 0; g < KSLICE / GSIZE; ++g) {          // 16 quant groups
        const int   gidx = g_begin + g;
        const float sc   = scale[row * NGROUP + gidx] * s2r;  // a
        const float zb   = -zero[row * NGROUP + gidx] * sc;   // b = -zero*a

        // Prefetch next group's weight stream (speculative; dropped if OOB).
        __builtin_prefetch(wrow + k_begin + (g + 1) * GSIZE + half * 8, 0, 1);

#pragma unroll
        for (int it = 0; it < 2; ++it) {                // 2 x K=32 WMMAs per group
            const int k0  = k_begin + g * GSIZE + it * 32;
            const int kb0 = k0 + half * 8;              // first 8-elem run
            // ---- B operand: 16 weights, two contiguous runs of 8 ints ----
            const int* wp = wrow + kb0;
            v4i q0 = *(const v4i*)(wp);
            v4i q1 = *(const v4i*)(wp + 4);
            v4i q2 = *(const v4i*)(wp + 16);
            v4i q3 = *(const v4i*)(wp + 20);
            // ---- A operand: 16 x values, same K offsets ----
            const float* xp = xrow + kb0;
            v4f x0 = *(const v4f*)(xp);
            v4f x1 = *(const v4f*)(xp + 4);
            v4f x2 = *(const v4f*)(xp + 16);
            v4f x3 = *(const v4f*)(xp + 20);

            v16bf A, B;
#pragma unroll
            for (int e = 0; e < 4; ++e) {
                B[e]      = (__bf16)__builtin_fmaf((float)q0[e], sc, zb);
                B[4 + e]  = (__bf16)__builtin_fmaf((float)q1[e], sc, zb);
                B[8 + e]  = (__bf16)__builtin_fmaf((float)q2[e], sc, zb);
                B[12 + e] = (__bf16)__builtin_fmaf((float)q3[e], sc, zb);
                A[e]      = (__bf16)x0[e];
                A[4 + e]  = (__bf16)x1[e];
                A[8 + e]  = (__bf16)x2[e];
                A[12 + e] = (__bf16)x3[e];
            }
            // D = A(16x32) x B(32x16) + C, f32 accumulate
            acc = __builtin_amdgcn_wmma_f32_16x16x32_bf16(
                /*neg_a=*/false, A, /*neg_b=*/false, B,
                /*c_mod=*/(short)0, acc, /*reuse_a=*/false, /*reuse_b=*/false);
        }
    }

    // ---- cross-wave K reduction via LDS ----
#pragma unroll
    for (int j = 0; j < 8; ++j) red[wave][lane][j] = acc[j];
    __syncthreads();

    if (wave == 0) {
        v8f tot = {};
#pragma unroll
        for (int w = 0; w < WAVES; ++w)
#pragma unroll
            for (int j = 0; j < 8; ++j) tot[j] += red[w][lane][j];

        const float bv = bias[obase + n];
        // C/D layout: VGPR j, lane -> token = j + 8*half, col = obase + n
#pragma unroll
        for (int j = 0; j < 8; ++j) {
            const int t = j + 8 * half;
            out[(long)t * OUTF + obase + n] = tot[j] + bv;
        }
    }
}

extern "C" void kernel_launch(void* const* d_in, const int* in_sizes, int n_in,
                              void* d_out, int out_size, void* d_ws, size_t ws_size,
                              hipStream_t stream) {
    const float* x     = (const float*)d_in[0];
    const int*   Wq    = (const int*)  d_in[1];
    const float* scale = (const float*)d_in[2];
    const float* zero  = (const float*)d_in[3];
    const float* s2    = (const float*)d_in[4];
    const float* bias  = (const float*)d_in[5];
    float*       out   = (float*)d_out;
    (void)in_sizes; (void)n_in; (void)out_size; (void)d_ws; (void)ws_size;

    dim3 grid(OUTF / 16);   // 512 blocks, one 16-feature tile each
    dim3 block(256);        // 8 wave32 waves, K split 8 ways
    sinq_wmma_kernel<<<grid, block, 0, stream>>>(x, Wq, scale, zero, s2, bias, out);
}
